// MultiNet_19902878450284
// MI455X (gfx1250) — matrix operand order repaired and checked by reading, not compiled, not run on hardware
//
#include <hip/hip_runtime.h>
#include <hip/hip_bf16.h>

typedef _Float16 half_t;
typedef _Float16 half8  __attribute__((ext_vector_type(8)));
typedef _Float16 half16 __attribute__((ext_vector_type(16)));
typedef float    float8 __attribute__((ext_vector_type(8)));

union H16 { half16 v; half8 h[2]; };

#define WMMA(a,b,c) __builtin_amdgcn_wmma_f32_16x16x32_f16(false,(a),false,(b),(short)0,(c),false,false)

#define EPSV 1e-4f
#define NPTS 100000

// ---------------- encode: feat0[y][x][c] = sdf[y][x]*enc_w[c] + enc_b[c] (f16 NHWC) ----------------
__global__ void k_encode(const float* __restrict__ sdf, const float* __restrict__ ew,
                         const float* __restrict__ eb, half_t* __restrict__ out) {
    int t = blockIdx.x * blockDim.x + threadIdx.x;     // 512*512*8 threads
    int pix = t >> 3;
    int c0  = (t & 7) * 8;
    float s = sdf[pix];
    half8 o;
    #pragma unroll
    for (int j = 0; j < 8; ++j) o[j] = (half_t)(s * ew[c0 + j] + eb[c0 + j]);
    *(half8*)(out + (size_t)pix * 64 + c0) = o;
}

// ---------------- conv weight repack: [l][oc][ic][ky][kx] f32 -> [l][tap][oc][ic] f16 ----------------
__global__ void k_convw(const float* __restrict__ w, half_t* __restrict__ out, int total) {
    int t = blockIdx.x * blockDim.x + threadIdx.x;
    if (t >= total) return;
    int kx = t % 5;  int r = t / 5;
    int ky = r % 5;  r /= 5;
    int ic = r % 64; r /= 64;
    int oc = r % 64; int l = r / 64;
    int tap = ky * 5 + kx;
    out[(((size_t)(l * 25 + tap) * 64 + oc) * 64) + ic] = (half_t)w[t];
}

__global__ void k_f32tof16(const float* __restrict__ in, half_t* __restrict__ out, int n) {
    int t = blockIdx.x * blockDim.x + threadIdx.x;
    if (t < n) out[t] = (half_t)in[t];
}

// ---------------- conv 5x5 same, 64->64, bias+relu: implicit GEMM via WMMA ----------------
// in/out NHWC f16 [S][S][64]; W f16 [tap][oc][ic]
// block = 8 waves = 64 oc x 64 px of one row; each wave: 2 oc-tiles x 1 px-tile (B reused 2x)
__global__ void __launch_bounds__(256) k_conv(const half_t* __restrict__ in, half_t* __restrict__ out,
                                              const half_t* __restrict__ W, const float* __restrict__ bias,
                                              int S) {
    __shared__ __align__(32) half_t tile[5 * 68 * 64];   // 43,520 B
    const int tid = threadIdx.x;
    const int tilesX = S >> 6;
    const int y  = blockIdx.x / tilesX;
    const int x0 = (blockIdx.x % tilesX) << 6;

    // stage 5 rows x 68 cols x 64 ch (zero-padded) into LDS, 16B chunks
    for (int t = tid; t < 2720; t += 256) {
        int pix = t >> 3, chunk = t & 7;
        int row = pix / 68, xx = pix - row * 68;
        int gy = y - 2 + row, gx = x0 - 2 + xx;
        half8 v = {};
        if ((unsigned)gy < (unsigned)S && (unsigned)gx < (unsigned)S)
            v = *(const half8*)(in + ((size_t)gy * S + gx) * 64 + chunk * 8);
        *(half8*)(tile + (size_t)(row * 68 + xx) * 64 + chunk * 8) = v;
    }
    __syncthreads();

    const int wave  = tid >> 5, lane = tid & 31;
    const int pxT   = wave & 3;                       // 4 pixel tiles
    const int oc0   = (wave >> 2) << 5;               // 2 oc-pair tiles: 0 or 32
    const int laneN = lane & 15;
    const int hi    = lane >> 4;

    float8 acc0 = {}, acc1 = {};
    const int arow = oc0 + laneN;                     // A rows: arow (tile0), arow+16 (tile1)
    #pragma unroll
    for (int tap = 0; tap < 25; ++tap) {
        int ky = tap / 5, kx = tap - ky * 5;
        const half_t* wrow0 = W + ((size_t)(tap * 64 + arow)) * 64;
        const half_t* wrow1 = wrow0 + (size_t)16 * 64;
        const half_t* brow  = tile + (size_t)(ky * 68 + (pxT * 16 + laneN + kx)) * 64;
        #pragma unroll
        for (int kk = 0; kk < 2; ++kk) {
            int ic0 = kk * 32;
            H16 a0, a1;
            a0.h[0] = *(const half8*)(wrow0 + ic0 + hi * 8);
            a0.h[1] = *(const half8*)(wrow0 + ic0 + 16 + hi * 8);
            a1.h[0] = *(const half8*)(wrow1 + ic0 + hi * 8);
            a1.h[1] = *(const half8*)(wrow1 + ic0 + 16 + hi * 8);
            H16 bb; bb.h[0] = *(const half8*)(brow + ic0 + hi * 8); bb.h[1] = *(const half8*)(brow + ic0 + 16 + hi * 8);    // table-order channels
            half16 b = bb.v;
            acc0 = WMMA(a0.v, b, acc0);
            acc1 = WMMA(a1.v, b, acc1);
        }
    }

    int px = x0 + pxT * 16 + laneN;
    half_t* obase = out + ((size_t)y * S + px) * 64;
    #pragma unroll
    for (int tt = 0; tt < 2; ++tt) {
        const float8& acc = tt ? acc1 : acc0;
        int ocb = oc0 + tt * 16 + hi * 8;
        half8 o;
        #pragma unroll
        for (int r = 0; r < 8; ++r) {
            float v = acc[r] + bias[ocb + r];
            o[r] = (half_t)(v > 0.f ? v : 0.f);
        }
        *(half8*)(obase + ocb) = o;
    }
}

// ---------------- 2x2 average pool, NHWC f16 ----------------
__global__ void k_pool(const half_t* __restrict__ in, half_t* __restrict__ out, int So) {
    int t = blockIdx.x * blockDim.x + threadIdx.x;
    int total = So * So * 8;
    if (t >= total) return;
    int c0 = (t & 7) * 8;
    int pix = t >> 3;
    int ox = pix % So, oy = pix / So;
    int Si = So * 2;
    const half_t* p = in + (((size_t)(oy * 2) * Si + ox * 2) * 64) + c0;
    half8 a = *(const half8*)p;
    half8 b = *(const half8*)(p + 64);
    half8 c = *(const half8*)(p + (size_t)Si * 64);
    half8 d = *(const half8*)(p + (size_t)Si * 64 + 64);
    half8 o;
    #pragma unroll
    for (int j = 0; j < 8; ++j)
        o[j] = (half_t)(0.25f * ((float)a[j] + (float)b[j] + (float)c[j] + (float)d[j]));
    *(half8*)(out + (size_t)pix * 64 + c0) = o;
}

// ---------------- bilinear sample 64ch into h[p][coff..coff+64] ----------------
__global__ void k_interp(const float* __restrict__ pts, const half_t* __restrict__ fmap,
                         int S, half_t* __restrict__ hout, int coff) {
    int t = blockIdx.x * blockDim.x + threadIdx.x;
    if (t >= NPTS * 8) return;
    int p  = t >> 3;
    int c0 = (t & 7) * 8;
    float px = pts[p * 2 + 0], py = pts[p * 2 + 1];
    px = fminf(fmaxf(px, EPSV), 1.f - EPSV);
    py = fminf(fmaxf(py, EPSV), 1.f - EPSV);
    float fx = px * (float)(S - 1), fy = py * (float)(S - 1);
    int xi = (int)floorf(fx); float xf = fx - (float)xi;
    int yi = (int)floorf(fy); float yf = fy - (float)yi;
    const half_t* base = fmap + (((size_t)yi * S + xi) * 64) + c0;
    half8 nw = *(const half8*)base;
    half8 ne = *(const half8*)(base + 64);
    half8 sw = *(const half8*)(base + (size_t)S * 64);
    half8 se = *(const half8*)(base + (size_t)S * 64 + 64);
    half8 o;
    #pragma unroll
    for (int j = 0; j < 8; ++j) {
        float n_ = ((float)ne[j] - (float)nw[j]) * xf + (float)nw[j];
        float s_ = ((float)se[j] - (float)sw[j]) * xf + (float)sw[j];
        o[j] = (half_t)((s_ - n_) * yf + n_);
    }
    *(half8*)(hout + (size_t)p * 256 + coff + c0) = o;
}

// ---------------- MLP layer: out[p][dout] = relu(h @ W^T + b), 2x2 WMMA register blocking ----------------
// wave computes 32 points x 32 dout (4 accumulators); fragments reused 2x each
__global__ void __launch_bounds__(256) k_mlp(const half_t* __restrict__ hin, const half_t* __restrict__ W,
                                             const float* __restrict__ bias, half_t* __restrict__ hout,
                                             int K, int dout) {
    int waveId = blockIdx.x * 8 + (threadIdx.x >> 5);
    int dT2 = dout >> 5;
    int ptT = waveId / dT2, ocT = waveId % dT2;
    int lane  = threadIdx.x & 31;
    int laneN = lane & 15;
    int hi    = lane >> 4;
    int oc0   = ocT * 32;
    int pt0   = ptT * 32;

    const half_t* arow0 = W   + (size_t)(oc0 + laneN) * K;
    const half_t* arow1 = arow0 + (size_t)16 * K;
    const half_t* brow0 = hin + (size_t)(pt0 + laneN) * K;
    const half_t* brow1 = brow0 + (size_t)16 * K;
    float8 acc00 = {}, acc01 = {}, acc10 = {}, acc11 = {};
    for (int k0 = 0; k0 < K; k0 += 32) {
        H16 a0, a1;
        a0.h[0] = *(const half8*)(arow0 + k0 + hi * 8);
        a0.h[1] = *(const half8*)(arow0 + k0 + 16 + hi * 8);
        a1.h[0] = *(const half8*)(arow1 + k0 + hi * 8);
        a1.h[1] = *(const half8*)(arow1 + k0 + 16 + hi * 8);
        H16 bb0, bb1;
        bb0.h[0] = *(const half8*)(brow0 + k0 + hi * 8);
        bb0.h[1] = *(const half8*)(brow0 + k0 + 16 + hi * 8);
        bb1.h[0] = *(const half8*)(brow1 + k0 + hi * 8);
        bb1.h[1] = *(const half8*)(brow1 + k0 + 16 + hi * 8);
        half16 b0 = bb0.v;
        half16 b1 = bb1.v;
        acc00 = WMMA(a0.v, b0, acc00);
        acc01 = WMMA(a0.v, b1, acc01);
        acc10 = WMMA(a1.v, b0, acc10);
        acc11 = WMMA(a1.v, b1, acc11);
    }
    #pragma unroll
    for (int pp = 0; pp < 2; ++pp) {
        int pt = pt0 + pp * 16 + laneN;
        half_t* obase = hout + (size_t)pt * dout;
        #pragma unroll
        for (int tt = 0; tt < 2; ++tt) {
            const float8& acc = pp ? (tt ? acc11 : acc01) : (tt ? acc10 : acc00);
            int ocb = oc0 + tt * 16 + hi * 8;
            half8 o;
            #pragma unroll
            for (int r = 0; r < 8; ++r) {
                float v = acc[r] + bias[ocb + r];
                o[r] = (half_t)(v > 0.f ? v : 0.f);
            }
            *(half8*)(obase + ocb) = o;
        }
    }
}

// ---------------- final 256 -> 1 in fp32 ----------------
__global__ void k_final(const half_t* __restrict__ h, const float* __restrict__ w,
                        const float* __restrict__ b, float* __restrict__ out) {
    int p = blockIdx.x * blockDim.x + threadIdx.x;
    if (p >= NPTS) return;
    const half_t* row = h + (size_t)p * 256;
    float acc = b[0];
    #pragma unroll 8
    for (int k = 0; k < 256; ++k) acc += (float)row[k] * w[k];
    out[p] = acc;
}

extern "C" void kernel_launch(void* const* d_in, const int* in_sizes, int n_in,
                              void* d_out, int out_size, void* d_ws, size_t ws_size,
                              hipStream_t stream) {
    const float* x      = (const float*)d_in[0];
    const float* sdf    = (const float*)d_in[1];
    const float* enc_w  = (const float*)d_in[2];
    const float* enc_b  = (const float*)d_in[3];
    const float* conv_w = (const float*)d_in[4];
    const float* conv_b = (const float*)d_in[5];
    const float* mlp_w0 = (const float*)d_in[6];   // (256,256)
    const float* mlp_w1 = (const float*)d_in[7];   // (512,256)
    const float* mlp_w2 = (const float*)d_in[8];   // (256,512)
    const float* mlp_w3 = (const float*)d_in[9];   // (1,256)
    const float* mlp_b0 = (const float*)d_in[10];
    const float* mlp_b1 = (const float*)d_in[11];
    const float* mlp_b2 = (const float*)d_in[12];
    const float* mlp_b3 = (const float*)d_in[13];

    char* ws = (char*)d_ws;
    const size_t FEAT = (size_t)512 * 512 * 64 * 2;          // 33,554,432 B
    half_t* featA = (half_t*)(ws);                           // encode / pooled inputs
    half_t* featB = (half_t*)(ws + FEAT);                    // conv outputs
    half_t* convW = (half_t*)(ws + 2 * FEAT);                // 819,200 B (pad to 1 MiB)
    half_t* mlpW0 = (half_t*)(ws + 2 * FEAT + (1u << 20));
    half_t* mlpW1 = mlpW0 + 256 * 256;
    half_t* mlpW2 = mlpW1 + 512 * 256;
    char*   hbase = ws + 2 * FEAT + (2u << 20);
    half_t* hR1 = (half_t*)hbase;                            // h0 (N*256) then h2 (N*512)
    half_t* hR2 = (half_t*)(hbase + (size_t)NPTS * 512 * 2); // h1 then h3 (N*256)
    float*  outp = (float*)d_out;

    // weight prep
    k_convw<<<(409600 + 255) / 256, 256, 0, stream>>>(conv_w, convW, 409600);
    k_f32tof16<<<(65536  + 255) / 256, 256, 0, stream>>>(mlp_w0, mlpW0, 65536);
    k_f32tof16<<<(131072 + 255) / 256, 256, 0, stream>>>(mlp_w1, mlpW1, 131072);
    k_f32tof16<<<(131072 + 255) / 256, 256, 0, stream>>>(mlp_w2, mlpW2, 131072);

    // encode
    k_encode<<<(512 * 512 * 8) / 256, 256, 0, stream>>>(sdf, enc_w, enc_b, featA);

    // conv / interp / pool pyramid
    for (int l = 0; l < 4; ++l) {
        int S = 512 >> l;
        k_conv<<<S * (S / 64), 256, 0, stream>>>(featA, featB,
                                                 convW + (size_t)l * 25 * 64 * 64,
                                                 conv_b + l * 64, S);
        k_interp<<<(NPTS * 8 + 255) / 256, 256, 0, stream>>>(x, featB, S, hR1, l * 64);
        if (l < 3) {
            int So = S / 2;
            k_pool<<<(So * So * 8 + 255) / 256, 256, 0, stream>>>(featB, featA, So);
        }
    }

    // MLP: h0(256)->h1(256)->h2(512)->h3(256)->out   (32x32 tiles per wave)
    const int ptT2 = NPTS / 32;                              // 3125, exact
    k_mlp<<<(ptT2 * 8)  / 8, 256, 0, stream>>>(hR1, mlpW0, mlp_b0, hR2, 256, 256); // h1
    k_mlp<<<(ptT2 * 16) / 8, 256, 0, stream>>>(hR2, mlpW1, mlp_b1, hR1, 256, 512); // h2 (h0 dead)
    k_mlp<<<(ptT2 * 8)  / 8, 256, 0, stream>>>(hR1, mlpW2, mlp_b2, hR2, 512, 256); // h3 (h1 dead)
    k_final<<<(NPTS + 255) / 256, 256, 0, stream>>>(hR2, mlp_w3, mlp_b3, outp);
}
